// GraphLayerConvSimple_51101520888322
// MI455X (gfx1250) — matrix operand order repaired and checked by reading, not compiled, not run on hardware
//
#include <hip/hip_runtime.h>

typedef __attribute__((ext_vector_type(16))) __bf16 v16bf;
typedef __attribute__((ext_vector_type(8)))  __bf16 v8bf;
typedef __attribute__((ext_vector_type(8)))  float  v8f;

#define TILE_CH   64
#define PADW      18
#define TILE_ELS  (PADW*PADW*TILE_CH)   /* 20736 bf16 = 41472 B */
#define TILE_DW   (TILE_ELS/2)          /* 10368 dwords */
#define PK_ELS    (4*9*2*512)           /* packed fragments per conv set */

__device__ __forceinline__ v16bf ld_frag16(const __bf16* p) {
  v16bf r;
  ((v8bf*)&r)[0] = ((const v8bf*)p)[0];
  ((v8bf*)&r)[1] = ((const v8bf*)p)[1];
  return r;
}

// A fragment: pre-packed weights, 32 contiguous bytes per lane
__device__ __forceinline__ v16bf load_a(const __bf16* __restrict__ pk, int mtile, int pos, int chunk, int lane) {
  return ld_frag16(pk + ((((mtile*9 + pos) << 1) + chunk) << 9) + lane*16);
}

// B fragment: shifted pixel from padded LDS tile; lane = column(N), lane>>4 selects K half
__device__ __forceinline__ v16bf load_b(const __bf16* tile, int ntile, int ky, int kx, int chunk, int lane) {
  int x = lane & 15, half = lane >> 4;
  int pix = (ntile + ky)*PADW + (x + kx);
  return ld_frag16(tile + pix*TILE_CH + chunk*32 + half*16);
}

__device__ __forceinline__ v8f vzero8() {
  v8f z; 
  #pragma unroll
  for (int e = 0; e < 8; ++e) z[e] = 0.0f;
  return z;
}

// ---------- weight packing: OIHW f32 -> bf16 WMMA-A fragment order ----------
__global__ __launch_bounds__(256) void pack_w(const float* __restrict__ src, int cin_total,
                                              int c_off, int n_valid, __bf16* __restrict__ dst) {
  int idx = blockIdx.x*256 + threadIdx.x;
  if (idx >= PK_ELS) return;
  int frag = idx >> 9, r = idx & 511;
  int lane = r >> 4, e = r & 15;
  int chunk = frag & 1;
  int tmp = frag >> 1;
  int pos = tmp % 9, mtile = tmp / 9;
  int m = mtile*16 + (lane & 15);
  int half = lane >> 4;
  int kl = half*16 + e;   // A-layout K index
  int cl = chunk*32 + kl;
  float v = (cl < n_valid) ? src[(m*cin_total + c_off + cl)*9 + pos] : 0.0f;
  dst[idx] = (__bf16)v;
}

// ---------- generic conv layer (one image per workgroup, 8 waves) ----------
// INMODE: 0 = map input (t-channel + 32 f32 channels, pad to 64), 1 = bf16 pixel-major [img][p][64]
// EPMODE: 0 = bias->bf16, 1 = bias+relu->bf16, 2 = bias->f32, 3 = aux*sigmoid(bias)->f32, 4 = raw->f32
template<int INMODE, int EPMODE>
__global__ __launch_bounds__(256) void conv_k(
    const __bf16* __restrict__ in_bf, const float* __restrict__ x_f,
    const float* __restrict__ tval, const __bf16* __restrict__ pk,
    const float* __restrict__ bias, const float* __restrict__ aux,
    float* __restrict__ out_f, __bf16* __restrict__ out_bf)
{
  __shared__ __align__(16) __bf16 sIn[TILE_ELS];
  const int tid = threadIdx.x;
  const int img = blockIdx.x;

  for (int i = tid; i < TILE_DW; i += 256) ((unsigned int*)sIn)[i] = 0u;
  __syncthreads();

  if (INMODE == 0) {
    int p = tid, y = p >> 4, x = p & 15;
    __bf16* d = sIn + ((y+1)*PADW + (x+1))*TILE_CH;
    d[0] = (__bf16)tval[0];
    #pragma unroll 8
    for (int c = 0; c < 32; ++c)
      d[1 + c] = (__bf16)x_f[(img*32 + c)*256 + p];
  } else {
    #pragma unroll
    for (int it = 0; it < 8; ++it) {
      int task = tid + it*256, pix = task >> 3, sub = task & 7;
      int y = pix >> 4, x = pix & 15;
      const uint4* s = (const uint4*)(in_bf + (size_t)(img*256 + pix)*TILE_CH + sub*8);
      *(uint4*)(sIn + ((y+1)*PADW + (x+1))*TILE_CH + sub*8) = *s;
    }
  }
  __syncthreads();

  const int lane = tid & 31, wave = tid >> 5;
  const int mtile = wave >> 1, nbase = (wave & 1)*8;
  const int half = lane >> 4;

  v8f acc[8];
  #pragma unroll
  for (int s = 0; s < 8; ++s) acc[s] = vzero8();

  for (int pos = 0; pos < 9; ++pos) {
    int ky = pos/3, kx = pos - ky*3;
    #pragma unroll
    for (int chunk = 0; chunk < 2; ++chunk) {
      v16bf a = load_a(pk, mtile, pos, chunk, lane);
      #pragma unroll
      for (int s = 0; s < 8; ++s) {
        v16bf b = load_b(sIn, nbase + s, ky, kx, chunk, lane);
        acc[s] = __builtin_amdgcn_wmma_f32_16x16x32_bf16(false, a, false, b, (short)0, acc[s], false, false);
      }
    }
  }

  float bv[8];
  #pragma unroll
  for (int r = 0; r < 8; ++r)
    bv[r] = (EPMODE == 4) ? 0.0f : bias[mtile*16 + half*8 + r];

  #pragma unroll
  for (int s = 0; s < 8; ++s) {
    int p = (nbase + s)*16 + (lane & 15);
    #pragma unroll
    for (int r = 0; r < 8; ++r) {
      int m = mtile*16 + half*8 + r;
      float v = acc[s][r] + bv[r];
      int idx = (img*256 + p)*TILE_CH + m;
      if (EPMODE == 0)      out_bf[idx] = (__bf16)v;
      else if (EPMODE == 1) out_bf[idx] = (__bf16)fmaxf(v, 0.0f);
      else if (EPMODE == 2) out_f[idx]  = v;
      else if (EPMODE == 3) out_f[idx]  = aux[idx] * (1.0f/(1.0f + __expf(-v)));
      else                  out_f[idx]  = v;
    }
  }
}

// ---------- pair-message kernel: one workgroup per (n, i), loop over j ----------
__global__ __launch_bounds__(512) void binary_k(
  const float* __restrict__ Pfix, const float* __restrict__ Psum,
  const float* __restrict__ Pgfix, const float* __restrict__ Pgsum,
  const float* __restrict__ b1, const float* __restrict__ bg1,
  const float* __restrict__ b2, const float* __restrict__ bg2,
  const __bf16* __restrict__ pkM, const __bf16* __restrict__ pkG,
  const float* __restrict__ u, float* __restrict__ out, int K)
{
  __shared__ __align__(16) __bf16 sM[TILE_ELS];
  __shared__ __align__(16) __bf16 sG[TILE_ELS];
  const int tid = threadIdx.x;
  const int imgI = blockIdx.x;
  const int n = imgI / K;
  const int base = n*K;

  for (int i = tid; i < TILE_DW; i += 512) { ((unsigned*)sM)[i] = 0u; ((unsigned*)sG)[i] = 0u; }

  const int sub = tid & 7;
  float bm[8], bgv[8];
  #pragma unroll
  for (int k = 0; k < 8; ++k) { bm[k] = b1[sub*8 + k]; bgv[k] = bg1[sub*8 + k]; }

  const int lane = tid & 31, wave = tid >> 5;
  const int mtile = wave >> 2, nbase = (wave & 3)*4;
  const int half = lane >> 4;
  float b2v[8], bg2v[8];
  #pragma unroll
  for (int r = 0; r < 8; ++r) { b2v[r] = b2[mtile*16 + half*8 + r]; bg2v[r] = bg2[mtile*16 + half*8 + r]; }

  v8f acc[4];
  #pragma unroll
  for (int s = 0; s < 4; ++s) acc[s] = vzero8();

  __syncthreads();

  for (int j = 0; j < K; ++j) {
    const int imgJ = base + j;
    // stage relu(Pfix_i + Psum_j + bias) as bf16 padded tiles (msg + gate)
    #pragma unroll
    for (int it = 0; it < 4; ++it) {
      int pix = (tid >> 3) + it*64;
      int y = pix >> 4, x = pix & 15;
      int tp = ((y+1)*PADW + (x+1))*TILE_CH + sub*8;
      const float* pf = Pfix  + (size_t)(imgI*256 + pix)*TILE_CH + sub*8;
      const float* ps = Psum  + (size_t)(imgJ*256 + pix)*TILE_CH + sub*8;
      const float* gf = Pgfix + (size_t)(imgI*256 + pix)*TILE_CH + sub*8;
      const float* gs = Pgsum + (size_t)(imgJ*256 + pix)*TILE_CH + sub*8;
      union { __bf16 h[8]; uint4 q; } vm, vg;
      #pragma unroll
      for (int k = 0; k < 8; ++k) {
        vm.h[k] = (__bf16)fmaxf(pf[k] + ps[k] + bm[k], 0.0f);
        vg.h[k] = (__bf16)fmaxf(gf[k] + gs[k] + bgv[k], 0.0f);
      }
      *(uint4*)(sM + tp) = vm.q;
      *(uint4*)(sG + tp) = vg.q;
    }
    __syncthreads();

    v8f dm[4], dg[4];
    #pragma unroll
    for (int s = 0; s < 4; ++s) { dm[s] = vzero8(); dg[s] = vzero8(); }

    for (int pos = 0; pos < 9; ++pos) {
      int ky = pos/3, kx = pos - ky*3;
      #pragma unroll
      for (int chunk = 0; chunk < 2; ++chunk) {
        v16bf am = load_a(pkM, mtile, pos, chunk, lane);
        #pragma unroll
        for (int s = 0; s < 4; ++s) {
          v16bf b = load_b(sM, nbase + s, ky, kx, chunk, lane);
          dm[s] = __builtin_amdgcn_wmma_f32_16x16x32_bf16(false, am, false, b, (short)0, dm[s], false, false);
        }
        v16bf ag = load_a(pkG, mtile, pos, chunk, lane);
        #pragma unroll
        for (int s = 0; s < 4; ++s) {
          v16bf b = load_b(sG, nbase + s, ky, kx, chunk, lane);
          dg[s] = __builtin_amdgcn_wmma_f32_16x16x32_bf16(false, ag, false, b, (short)0, dg[s], false, false);
        }
      }
    }

    #pragma unroll
    for (int s = 0; s < 4; ++s) {
      #pragma unroll
      for (int r = 0; r < 8; ++r) {
        float msg  = dm[s][r] + b2v[r];
        float gate = 1.0f/(1.0f + __expf(-(dg[s][r] + bg2v[r])));
        acc[s][r] += msg*gate;
      }
    }
    __syncthreads();
  }

  const float inv = 1.0f / (float)(K - 1);
  #pragma unroll
  for (int s = 0; s < 4; ++s) {
    int p = (nbase + s)*16 + (lane & 15);
    #pragma unroll
    for (int r = 0; r < 8; ++r) {
      int m = mtile*16 + half*8 + r;
      float v = u[(size_t)(imgI*256 + p)*TILE_CH + m] + acc[s][r]*inv;
      out[(size_t)(imgI*TILE_CH + m)*256 + p] = v;   // NCHW output layout
    }
  }
}

extern "C" void kernel_launch(void* const* d_in, const int* in_sizes, int n_in,
                              void* d_out, int out_size, void* d_ws, size_t ws_size,
                              hipStream_t stream) {
  (void)in_sizes; (void)n_in; (void)out_size; (void)ws_size;
  const float* t     = (const float*)d_in[0];
  const float* x     = (const float*)d_in[1];
  const float* w_map = (const float*)d_in[2];
  const float* b_map = (const float*)d_in[3];
  const float* w_u1  = (const float*)d_in[4];
  const float* b_u1  = (const float*)d_in[5];
  const float* w_u2  = (const float*)d_in[6];
  const float* b_u2  = (const float*)d_in[7];
  const float* w_ug1 = (const float*)d_in[8];
  const float* b_ug1 = (const float*)d_in[9];
  const float* w_ug2 = (const float*)d_in[10];
  const float* b_ug2 = (const float*)d_in[11];
  const float* w_b1  = (const float*)d_in[12];
  const float* b_b1  = (const float*)d_in[13];
  const float* w_b2  = (const float*)d_in[14];
  const float* b_b2  = (const float*)d_in[15];
  const float* w_bg1 = (const float*)d_in[16];
  const float* b_bg1 = (const float*)d_in[17];
  const float* w_bg2 = (const float*)d_in[18];
  const float* b_bg2 = (const float*)d_in[19];

  const int NK = 40, K = 20;
  const size_t IMG_BF = (size_t)NK*256*64*sizeof(__bf16);
  const size_t IMG_F  = (size_t)NK*256*64*sizeof(float);
  const size_t PKB    = (size_t)PK_ELS*sizeof(__bf16);

  char* ws = (char*)d_ws;
  size_t off = 0;
  auto carve = [&](size_t bytes) -> void* {
    void* p = ws + off;
    off = (off + bytes + 255) & ~(size_t)255;
    return p;
  };
  __bf16* pk_map  = (__bf16*)carve(PKB);
  __bf16* pk_u1   = (__bf16*)carve(PKB);
  __bf16* pk_u2   = (__bf16*)carve(PKB);
  __bf16* pk_ug1  = (__bf16*)carve(PKB);
  __bf16* pk_ug2  = (__bf16*)carve(PKB);
  __bf16* pk_b1a  = (__bf16*)carve(PKB);
  __bf16* pk_b1b  = (__bf16*)carve(PKB);
  __bf16* pk_bg1a = (__bf16*)carve(PKB);
  __bf16* pk_bg1b = (__bf16*)carve(PKB);
  __bf16* pk_b2   = (__bf16*)carve(PKB);
  __bf16* pk_bg2  = (__bf16*)carve(PKB);
  __bf16* h_bf    = (__bf16*)carve(IMG_BF);
  __bf16* u1a     = (__bf16*)carve(IMG_BF);
  __bf16* g1a     = (__bf16*)carve(IMG_BF);
  float*  uLin    = (float*)carve(IMG_F);
  float*  uacc    = (float*)carve(IMG_F);
  float*  Pf      = (float*)carve(IMG_F);
  float*  Ps      = (float*)carve(IMG_F);
  float*  Pgf     = (float*)carve(IMG_F);
  float*  Pgs     = (float*)carve(IMG_F);

  const int PG = (PK_ELS + 255)/256;
  pack_w<<<PG,256,0,stream>>>(w_map,  33,  0, 33, pk_map);
  pack_w<<<PG,256,0,stream>>>(w_u1,   64,  0, 64, pk_u1);
  pack_w<<<PG,256,0,stream>>>(w_u2,   64,  0, 64, pk_u2);
  pack_w<<<PG,256,0,stream>>>(w_ug1,  64,  0, 64, pk_ug1);
  pack_w<<<PG,256,0,stream>>>(w_ug2,  64,  0, 64, pk_ug2);
  pack_w<<<PG,256,0,stream>>>(w_b1,  128,  0, 64, pk_b1a);   // applied to h_i (fixed, first concat half)
  pack_w<<<PG,256,0,stream>>>(w_b1,  128, 64, 64, pk_b1b);   // applied to h_j (summed, second half)
  pack_w<<<PG,256,0,stream>>>(w_bg1, 128,  0, 64, pk_bg1a);
  pack_w<<<PG,256,0,stream>>>(w_bg1, 128, 64, 64, pk_bg1b);
  pack_w<<<PG,256,0,stream>>>(w_b2,   64,  0, 64, pk_b2);
  pack_w<<<PG,256,0,stream>>>(w_bg2,  64,  0, 64, pk_bg2);

  // h = conv(concat(t, x))
  conv_k<0,0><<<NK,256,0,stream>>>(nullptr, x, t, pk_map, b_map, nullptr, nullptr, h_bf);
  // unary path
  conv_k<1,1><<<NK,256,0,stream>>>(h_bf, nullptr, nullptr, pk_u1,  b_u1,  nullptr, nullptr, u1a);
  conv_k<1,2><<<NK,256,0,stream>>>(u1a,  nullptr, nullptr, pk_u2,  b_u2,  nullptr, uLin,    nullptr);
  conv_k<1,1><<<NK,256,0,stream>>>(h_bf, nullptr, nullptr, pk_ug1, b_ug1, nullptr, nullptr, g1a);
  conv_k<1,3><<<NK,256,0,stream>>>(g1a,  nullptr, nullptr, pk_ug2, b_ug2, uLin,    uacc,    nullptr);
  // factorized first pair-conv (linear split of the channel concat)
  conv_k<1,4><<<NK,256,0,stream>>>(h_bf, nullptr, nullptr, pk_b1a,  nullptr, nullptr, Pf,  nullptr);
  conv_k<1,4><<<NK,256,0,stream>>>(h_bf, nullptr, nullptr, pk_b1b,  nullptr, nullptr, Ps,  nullptr);
  conv_k<1,4><<<NK,256,0,stream>>>(h_bf, nullptr, nullptr, pk_bg1a, nullptr, nullptr, Pgf, nullptr);
  conv_k<1,4><<<NK,256,0,stream>>>(h_bf, nullptr, nullptr, pk_bg1b, nullptr, nullptr, Pgs, nullptr);
  // per-pair second convs + gate + reduction over j, fused with unary add
  binary_k<<<NK,512,0,stream>>>(Pf, Ps, Pgf, Pgs, b_b1, b_bg1, b_b2, b_bg2,
                                pk_b2, pk_bg2, uacc, (float*)d_out, K);
}
